// PyramidAttention_2095944040858
// MI455X (gfx1250) — matrix operand order repaired and checked by reading, not compiled, not run on hardware
//
#include <hip/hip_runtime.h>
#include <cstdint>
#include <cstddef>

// ---------------------------------------------------------------------------
// PyramidAttention on MI455X (gfx1250): fused flash-attention formulation.
//   Q [B,2304,288] f16  = im2col(prelu(conv1x1(x,Wb)))
//   K [B,NP,288]  f16   = per-patch-normalized mat patches (5 scales), NP=7552
//   Vt[B,576,NP]  f16   = base patches, stored transposed for B-fragment loads
//   P [B,2304,576] f32  = softmax(10*Q K^T) V      (online softmax, WMMA f16)
//   out = x + 0.25 * fold3x3(P)
//
// Flash kernel: 4-wave workgroup, N advances 128/iteration. Each wave owns a
// 32-wide score subtile (no redundant score WMMAs, K loaded once per block)
// and a 144-wide V slice for P@V. 54 static v_wmma sites.
// ---------------------------------------------------------------------------

#define BATCH   4
#define CIN     64
#define HH      48
#define WW      48
#define HWP     (HH * WW)      // 2304
#define DK      288            // 32*3*3
#define DV      576            // 64*3*3
#define NREAL   7470           // 2304+1849+1444+1089+784
#define NPAD    7552           // multiple of 128
#define SMSCALE 10.0f

typedef _Float16 v16h __attribute__((ext_vector_type(16)));
typedef float    v8f  __attribute__((ext_vector_type(8)));

union FragA { v16h v; uint32_t u[8]; uint4 q[2]; _Float16 h[16]; };
union FragC { v8f  v; float f[8]; };

__device__ __forceinline__ int clampi(int v, int lo, int hi) {
    return v < lo ? lo : (v > hi ? hi : v);
}

// ---------------------------------------------------------------------------
// 1x1 conv + PReLU (generic): out[b,co,p] = prelu(sum_ci W[co,ci]*in[b,ci,p]+bias)
// ---------------------------------------------------------------------------
__global__ void conv1x1_prelu_kernel(const float* __restrict__ in,
                                     const float* __restrict__ W,
                                     const float* __restrict__ bias,
                                     const float* __restrict__ alpha,
                                     float* __restrict__ out,
                                     int Cout, int HWs) {
    int idx = blockIdx.x * blockDim.x + threadIdx.x;
    int total = BATCH * Cout * HWs;
    if (idx >= total) return;
    int p  = idx % HWs;
    int t  = idx / HWs;
    int co = t % Cout;
    int b  = t / Cout;
    const float* inb = in + (size_t)b * CIN * HWs + p;
    const float* w   = W + (size_t)co * CIN;
    float acc = bias[co];
    #pragma unroll 8
    for (int ci = 0; ci < CIN; ++ci) acc += w[ci] * inb[(size_t)ci * HWs];
    float a = alpha[0];
    out[idx] = acc >= 0.0f ? acc : a * acc;
}

// ---------------------------------------------------------------------------
// Bilinear resize (half-pixel centers, edge clamp) x[B,64,48,48] -> [B,64,Hs,Ws]
// ---------------------------------------------------------------------------
__global__ void resize_kernel(const float* __restrict__ x,
                              float* __restrict__ out, int Hs, int Ws) {
    int idx = blockIdx.x * blockDim.x + threadIdx.x;
    int total = BATCH * CIN * Hs * Ws;
    if (idx >= total) return;
    int xs = idx % Ws;
    int t  = idx / Ws;
    int ys = t % Hs;
    int bc = t / Hs;                       // b*64 + c
    float fy = (ys + 0.5f) * ((float)HH / (float)Hs) - 0.5f;
    float fx = (xs + 0.5f) * ((float)WW / (float)Ws) - 0.5f;
    int y0 = (int)floorf(fy); float wy = fy - (float)y0;
    int x0 = (int)floorf(fx); float wx = fx - (float)x0;
    int y0c = clampi(y0, 0, HH - 1), y1c = clampi(y0 + 1, 0, HH - 1);
    int x0c = clampi(x0, 0, WW - 1), x1c = clampi(x0 + 1, 0, WW - 1);
    const float* xb = x + (size_t)bc * HWP;
    float v00 = xb[y0c * WW + x0c], v01 = xb[y0c * WW + x1c];
    float v10 = xb[y1c * WW + x0c], v11 = xb[y1c * WW + x1c];
    out[idx] = v00 * (1.f - wy) * (1.f - wx) + v01 * (1.f - wy) * wx
             + v10 * wy * (1.f - wx)         + v11 * wy * wx;
}

// ---------------------------------------------------------------------------
// Q build: Q[b, p, d=c*9+kh*3+kw] = mb[b,c, y+kh-1, x+kw-1] (zero pad), f16
// ---------------------------------------------------------------------------
__global__ void build_q_kernel(const float* __restrict__ mb,
                               _Float16* __restrict__ Q) {
    int idx = blockIdx.x * blockDim.x + threadIdx.x;
    int total = BATCH * HWP * DK;
    if (idx >= total) return;
    int d = idx % DK;
    int t = idx / DK;
    int p = t % HWP;
    int b = t / HWP;
    int c = d / 9, k = d % 9, kh = k / 3, kw = k % 3;
    int y = p / WW + kh - 1, x = p % WW + kw - 1;
    float v = (y >= 0 && y < HH && x >= 0 && x < WW)
            ? mb[((size_t)(b * 32 + c)) * HWP + y * WW + x] : 0.0f;
    Q[idx] = (_Float16)v;
}

// ---------------------------------------------------------------------------
// K/V build for one scale: per-patch L2-normalized mat patch -> K row (f16),
// base patch -> Vt column (transposed, f16).
// ---------------------------------------------------------------------------
__global__ void build_kv_kernel(const float* __restrict__ mat,
                                const float* __restrict__ base,
                                _Float16* __restrict__ K,
                                _Float16* __restrict__ Vt,
                                int n0, int Hs, int Ws) {
    int HWs = Hs * Ws;
    int idx = blockIdx.x * blockDim.x + threadIdx.x;
    if (idx >= BATCH * HWs) return;
    int p = idx % HWs;
    int b = idx / HWs;
    int py = p / Ws, px = p % Ws;

    const float* matb = mat + (size_t)b * 32 * HWs;
    float ss = 0.0f;
    for (int c = 0; c < 32; ++c)
        for (int kh = 0; kh < 3; ++kh)
            for (int kw = 0; kw < 3; ++kw) {
                int yy = py + kh - 1, xx = px + kw - 1;
                float v = (yy >= 0 && yy < Hs && xx >= 0 && xx < Ws)
                        ? matb[(size_t)c * HWs + yy * Ws + xx] : 0.0f;
                ss += v * v;
            }
    float inv = 1.0f / fmaxf(sqrtf(ss), 1e-4f);

    _Float16* krow = K + ((size_t)b * NPAD + n0 + p) * DK;
    int d = 0;
    for (int c = 0; c < 32; ++c)
        for (int kh = 0; kh < 3; ++kh)
            for (int kw = 0; kw < 3; ++kw) {
                int yy = py + kh - 1, xx = px + kw - 1;
                float v = (yy >= 0 && yy < Hs && xx >= 0 && xx < Ws)
                        ? matb[(size_t)c * HWs + yy * Ws + xx] : 0.0f;
                krow[d++] = (_Float16)(v * inv);
            }

    const float* baseb = base + (size_t)b * 64 * HWs;
    _Float16* vtb = Vt + (size_t)b * DV * NPAD;
    d = 0;
    for (int c = 0; c < 64; ++c)
        for (int kh = 0; kh < 3; ++kh)
            for (int kw = 0; kw < 3; ++kw) {
                int yy = py + kh - 1, xx = px + kw - 1;
                float v = (yy >= 0 && yy < Hs && xx >= 0 && xx < Ws)
                        ? baseb[(size_t)c * HWs + yy * Ws + xx] : 0.0f;
                vtb[(size_t)d * NPAD + n0 + p] = (_Float16)v;
                ++d;
            }
}

// ---------------------------------------------------------------------------
// Flash attention: 4 wave32 per block (128 threads).
//   blockIdx.x = rowTile (0..143), blockIdx.y = batch.
//   Wave w: computes score subtile for N columns [jb+32w, jb+32w+32) into LDS
//   (no redundant score WMMAs), then P@V for V-column slice [144w, 144w+144).
//   N tile per iteration: 128. LDS rows padded to dodge bank conflicts.
// ---------------------------------------------------------------------------
#define SS_STRIDE 132   // floats  (132 % 64 == 4 -> distinct banks per row)
#define SP_STRIDE 136   // halves  (272B row stride -> distinct banks per row)

__global__ __launch_bounds__(128)
void flash_attn_kernel(const _Float16* __restrict__ Q,
                       const _Float16* __restrict__ K,
                       const _Float16* __restrict__ Vt,
                       float* __restrict__ P) {
    __shared__ __align__(16) float    sS[16][SS_STRIDE];
    __shared__ __align__(16) _Float16 sP[16][SP_STRIDE];
    __shared__ __align__(16) float    sScale[16];
    __shared__ __align__(16) float    sL[16];

    const int lane = threadIdx.x & 31;
    const int wave = threadIdx.x >> 5;       // 0..3
    const int half = lane >> 4;
    const int nn   = lane & 15;
    const int bi      = blockIdx.y;
    const int rowBase = blockIdx.x * 16;
    const int colBase = wave * 144;          // this wave's V slice
    const int jsub    = wave * 32;           // this wave's score subtile
    const bool writer = (wave == 0) && (half == 0);

    const _Float16* Qb = Q  + (size_t)bi * HWP * DK;
    const _Float16* Kb = K  + (size_t)bi * NPAD * DK;
    const _Float16* Vb = Vt + (size_t)bi * DV * NPAD;
    float*          Pb = P  + (size_t)bi * HWP * DV;

    // --- resident Q A-fragments: lane holds row M=nn, pair pattern over K ---
    FragA qa[9];
    {
        const _Float16* qrow = Qb + (size_t)(rowBase + nn) * DK;
        #pragma unroll
        for (int f = 0; f < 9; ++f) {
            #pragma unroll
            for (int v = 0; v < 8; ++v) {
                int kp = 2 * v + 16 * half;
                qa[f].u[v] = *reinterpret_cast<const uint32_t*>(qrow + f * 32 + kp);
            }
        }
    }

    FragC o[9];
    #pragma unroll
    for (int f = 0; f < 9; ++f)
        #pragma unroll
        for (int r = 0; r < 8; ++r) o[f].f[r] = 0.0f;

    float m = -3.0e38f, l = 0.0f;   // running stats for row nn (dup per lane-group)

    for (int jb = 0; jb < NPAD; jb += 128) {
        if (jb + 128 < NPAD)        // prefetch this wave's next K subtile
            __builtin_prefetch(Kb + (size_t)(jb + 128 + jsub + lane) * DK, 0, 1);

        // ---- S subtile = Q @ K[jb+jsub : jb+jsub+32]^T : two D fragments ----
        FragC s0, s1;
        #pragma unroll
        for (int r = 0; r < 8; ++r) { s0.f[r] = 0.0f; s1.f[r] = 0.0f; }
        #pragma unroll
        for (int f = 0; f < 9; ++f) {
            FragA kb0, kb1;
            const _Float16* k0 = Kb + (size_t)(jb + jsub +      nn) * DK + f * 32 + 16 * half;
            const _Float16* k1 = Kb + (size_t)(jb + jsub + 16 + nn) * DK + f * 32 + 16 * half;
            kb0.q[0] = reinterpret_cast<const uint4*>(k0)[0];
            kb0.q[1] = reinterpret_cast<const uint4*>(k0)[1];
            kb1.q[0] = reinterpret_cast<const uint4*>(k1)[0];
            kb1.q[1] = reinterpret_cast<const uint4*>(k1)[1];
            s0.v = __builtin_amdgcn_wmma_f32_16x16x32_f16(false, qa[f].v, false, kb0.v,
                                                          (short)0, s0.v, false, false);
            s1.v = __builtin_amdgcn_wmma_f32_16x16x32_f16(false, qa[f].v, false, kb1.v,
                                                          (short)0, s1.v, false, false);
        }

        // ---- scale + pad-mask, stage in LDS (row = r+8*half, col = jsub+nn) ----
        {
            int j0 = jb + jsub + nn, j1 = jb + jsub + 16 + nn;
            #pragma unroll
            for (int r = 0; r < 8; ++r) {
                int row = r + 8 * half;
                sS[row][jsub + nn]      = (j0 < NREAL) ? s0.f[r] * SMSCALE : -1.0e30f;
                sS[row][jsub + nn + 16] = (j1 < NREAL) ? s1.f[r] * SMSCALE : -1.0e30f;
            }
        }
        __syncthreads();

        // ---- online softmax for row nn over 128 columns ----
        {
            float tmax = -3.0e38f;
            #pragma unroll 8
            for (int c = 0; c < 128; ++c) tmax = fmaxf(tmax, sS[nn][c]);
            float mnew  = fmaxf(m, tmax);
            float scale = __expf(m - mnew);
            float sum = 0.0f;
            #pragma unroll 8
            for (int c = 0; c < 128; ++c) {
                float pv = __expf(sS[nn][c] - mnew);
                sum += pv;
                if (writer) sP[nn][c] = (_Float16)pv;
            }
            if (writer) sScale[nn] = scale;
            l = l * scale + sum;
            m = mnew;
        }
        __syncthreads();

        // ---- rescale accumulators by per-row correction ----
        #pragma unroll
        for (int r = 0; r < 8; ++r) {
            float sc = sScale[r + 8 * half];
            #pragma unroll
            for (int f = 0; f < 9; ++f) o[f].f[r] *= sc;
        }

        // ---- O += P @ V over the 4 sub-tiles, this wave's 144 columns ----
        #pragma unroll
        for (int t = 0; t < 4; ++t) {
            FragA pa;
            #pragma unroll
            for (int v = 0; v < 8; ++v) {
                int kp = 2 * v + 16 * half;
                pa.u[v] = *reinterpret_cast<const uint32_t*>(&sP[nn][t * 32 + kp]);
            }
            #pragma unroll
            for (int f = 0; f < 9; ++f) {
                FragA vb;
                const _Float16* vp = Vb + (size_t)(colBase + f * 16 + nn) * NPAD
                                        + jb + t * 32 + 16 * half;
                vb.q[0] = reinterpret_cast<const uint4*>(vp)[0];
                vb.q[1] = reinterpret_cast<const uint4*>(vp)[1];
                o[f].v = __builtin_amdgcn_wmma_f32_16x16x32_f16(false, pa.v, false, vb.v,
                                                                (short)0, o[f].v, false, false);
            }
        }
        __syncthreads();    // sS/sP reused next iteration
    }

    // ---- finalize: divide by row sums, write P ----
    if (writer) sL[nn] = l;
    __syncthreads();
    #pragma unroll
    for (int r = 0; r < 8; ++r) {
        float inv = 1.0f / sL[r + 8 * half];
        int row = rowBase + r + 8 * half;
        #pragma unroll
        for (int f = 0; f < 9; ++f)
            Pb[(size_t)row * DV + colBase + f * 16 + nn] = o[f].f[r] * inv;
    }
}

// ---------------------------------------------------------------------------
// col2im fold (conv_transpose2d, pad 1) + /4 + residual
// ---------------------------------------------------------------------------
__global__ void fold_kernel(const float* __restrict__ P,
                            const float* __restrict__ x,
                            float* __restrict__ out) {
    int idx = blockIdx.x * blockDim.x + threadIdx.x;
    int total = BATCH * 64 * HWP;
    if (idx >= total) return;
    int p = idx % HWP;
    int t = idx / HWP;
    int c = t % 64;
    int b = t / 64;
    int y = p / WW, xq = p % WW;
    const float* Pb = P + (size_t)b * HWP * DV;
    float acc = 0.0f;
    #pragma unroll
    for (int u = 0; u < 3; ++u)
        #pragma unroll
        for (int v = 0; v < 3; ++v) {
            int py = y + 1 - u, px = xq + 1 - v;
            if (py >= 0 && py < HH && px >= 0 && px < WW)
                acc += Pb[(size_t)(py * WW + px) * DV + c * 9 + u * 3 + v];
        }
    out[idx] = x[idx] + 0.25f * acc;
}

// ---------------------------------------------------------------------------
extern "C" void kernel_launch(void* const* d_in, const int* in_sizes, int n_in,
                              void* d_out, int out_size, void* d_ws, size_t ws_size,
                              hipStream_t stream) {
    const float* x  = (const float*)d_in[0];
    const float* Wb = (const float*)d_in[1];
    const float* bb = (const float*)d_in[2];
    const float* ab = (const float*)d_in[3];
    const float* Wm = (const float*)d_in[4];
    const float* bm = (const float*)d_in[5];
    const float* am = (const float*)d_in[6];
    const float* Wa = (const float*)d_in[7];
    const float* ba = (const float*)d_in[8];
    const float* aa = (const float*)d_in[9];
    float* out = (float*)d_out;

    // Workspace bump allocator (all offsets 256B aligned). Total ~87 MB.
    char* ws = (char*)d_ws;
    size_t off = 0;
    auto bump = [&](size_t bytes) {
        size_t o = off;
        off = (off + bytes + 255) & ~(size_t)255;
        return o;
    };
    float*    mb    = (float*)   (ws + bump((size_t)BATCH * 32 * HWP * 4));
    _Float16* Q     = (_Float16*)(ws + bump((size_t)BATCH * HWP * DK * 2));
    _Float16* Kmat  = (_Float16*)(ws + bump((size_t)BATCH * NPAD * DK * 2));
    _Float16* Vt    = (_Float16*)(ws + bump((size_t)BATCH * DV * NPAD * 2));
    float*    Pbuf  = (float*)   (ws + bump((size_t)BATCH * HWP * DV * 4));
    float*    xr    = (float*)   (ws + bump((size_t)BATCH * CIN * HWP * 4));
    float*    matb  = (float*)   (ws + bump((size_t)BATCH * 32 * HWP * 4));
    float*    baseb = (float*)   (ws + bump((size_t)BATCH * 64 * HWP * 4));
    (void)ws_size; (void)in_sizes; (void)n_in; (void)out_size;

    const int TPB = 256;
    auto blocks = [](long long n, int t) { return (int)((n + t - 1) / t); };

    // match_base = prelu(conv1x1(x, Wb, bb), ab)
    conv1x1_prelu_kernel<<<blocks((long long)BATCH * 32 * HWP, TPB), TPB, 0, stream>>>(
        x, Wb, bb, ab, mb, 32, HWP);
    // Q = im2col(match_base), f16
    build_q_kernel<<<blocks((long long)BATCH * HWP * DK, TPB), TPB, 0, stream>>>(mb, Q);

    const int hs[5]  = {48, 43, 38, 33, 28};
    const int n0s[5] = {0, 2304, 4153, 5597, 6686};
    for (int si = 0; si < 5; ++si) {
        int Hs = hs[si], Wsz = hs[si], HWs = Hs * Wsz;
        resize_kernel<<<blocks((long long)BATCH * CIN * HWs, TPB), TPB, 0, stream>>>(
            x, xr, Hs, Wsz);
        conv1x1_prelu_kernel<<<blocks((long long)BATCH * 32 * HWs, TPB), TPB, 0, stream>>>(
            xr, Wm, bm, am, matb, 32, HWs);
        conv1x1_prelu_kernel<<<blocks((long long)BATCH * 64 * HWs, TPB), TPB, 0, stream>>>(
            xr, Wa, ba, aa, baseb, 64, HWs);
        build_kv_kernel<<<blocks((long long)BATCH * HWs, TPB), TPB, 0, stream>>>(
            matb, baseb, Kmat, Vt, n0s[si], Hs, Wsz);
    }

    // fused attention: 144 row tiles x 4 batches, 4 waves (128 thr) per block
    flash_attn_kernel<<<dim3(144, BATCH), 128, 0, stream>>>(Q, Kmat, Vt, Pbuf);

    // fold + residual
    fold_kernel<<<blocks((long long)BATCH * 64 * HWP, TPB), TPB, 0, stream>>>(Pbuf, x, out);
}
